// GAT_51917564674531
// MI455X (gfx1250) — matrix operand order repaired and checked by reading, not compiled, not run on hardware
//
#include <hip/hip_runtime.h>

// ---------------------------------------------------------------------------
// Types for CDNA5 WMMA (gfx1250, wave32)
// ---------------------------------------------------------------------------
typedef __attribute__((ext_vector_type(16))) __bf16 v16bf;
typedef __attribute__((ext_vector_type(8)))  __bf16 bf16x8;
typedef __attribute__((ext_vector_type(8)))  float  v8f;
typedef int v4i __attribute__((vector_size(16)));

#define F_IN   2048
#define H1     512
#define C_HEAD 256

// GEMM tiling
#define BM 128
#define BN 128
#define BK 64
#define LDS_STRIDE 72   // 64 + 8 bf16 pad: row stride 144B = 36 banks -> conflict-free

// gfx1250 async global->LDS path (guarded; falls back to load+ds_store)
#if defined(__has_builtin)
#if __has_builtin(__builtin_amdgcn_global_load_async_to_lds_b128) && \
    __has_builtin(__builtin_amdgcn_s_wait_asynccnt)
#define USE_ASYNC_B 1
#endif
#endif

// fp32 -> bf16 round-to-nearest-even
__device__ __forceinline__ unsigned short f2bf(float f) {
  unsigned u = __float_as_uint(f);
  u += 0x7FFFu + ((u >> 16) & 1u);
  return (unsigned short)(u >> 16);
}
__device__ __forceinline__ unsigned pack2bf(float a, float b) {
#if defined(__has_builtin) && __has_builtin(__builtin_amdgcn_cvt_pk_bf16_f32)
  typedef __attribute__((ext_vector_type(2))) __bf16 bf16x2;
  union { bf16x2 v; unsigned u; } cv;
  cv.v = __builtin_amdgcn_cvt_pk_bf16_f32(a, b);
  return cv.u;
#else
  return (unsigned)f2bf(a) | ((unsigned)f2bf(b) << 16);
#endif
}

// monotone float<->uint mapping so atomicMax(uint) == float max
__device__ __forceinline__ unsigned f2ord(float f) {
  unsigned u = __float_as_uint(f);
  return u ^ ((u >> 31) ? 0xFFFFFFFFu : 0x80000000u);
}
__device__ __forceinline__ float ord2f(unsigned u) {
  unsigned v = u ^ ((u >> 31) ? 0x80000000u : 0xFFFFFFFFu);
  return __uint_as_float(v);
}

// ---------------------------------------------------------------------------
// Weight pre-pass: fp32 [K][N] row-major  ->  bf16 [N][K] (transposed)
// ---------------------------------------------------------------------------
__global__ void convert_transpose_kernel(const float* __restrict__ W,
                                         unsigned short* __restrict__ Wt,
                                         int K, int N) {
  long long idx = (long long)blockIdx.x * blockDim.x + threadIdx.x;
  if (idx >= (long long)K * N) return;
  int n = (int)(idx % N);
  int k = (int)(idx / N);
  Wt[(size_t)n * K + k] = f2bf(W[idx]);
}

// ---------------------------------------------------------------------------
// Tiled bf16 WMMA GEMM:  C[M,N] = A[M,K] @ Bt[N,K]^T
//   A : fp32 row-major (converted to bf16 while staging, b128 granular)
//   Bt: bf16, N-major (pre-transposed weights; async-copied to LDS)
// Block: 256 threads (8 waves, 4Mx2N). Block tile 128x128, K-step 64.
// Each wave: 32x64 tile = 8 f32 accumulators, 16 WMMAs per K-step.
// ---------------------------------------------------------------------------
__global__ __launch_bounds__(256)
void gemm_bf16_wmma_kernel(const float* __restrict__ A,
                           const unsigned short* __restrict__ Bt,
                           float* __restrict__ C,
                           int M, int N, int K)
{
  __shared__ unsigned short As[BM * LDS_STRIDE];   // 18 KB
  __shared__ unsigned short Bs[BN * LDS_STRIDE];   // 18 KB

  const int tid   = threadIdx.x;
  const int lane  = tid & 31;
  const int wave  = tid >> 5;
  const int waveM = wave >> 1;          // 0..3  -> 32-row sub-tile
  const int waveN = wave & 1;           // 0..1  -> 64-col sub-tile
  const int blockM = blockIdx.y * BM;
  const int blockN = blockIdx.x * BN;
  const int half = lane >> 4;           // lane half selects K sub-range
  const int l16  = lane & 15;
  const bool full = (blockM + BM <= M); // uniform: 156/157 blocks take fast path

  v8f acc[2][4] = {};

  for (int k0 = 0; k0 < K; k0 += BK) {
    // ---- stage A (128x64 fp32 -> bf16): 4 chunks of 8 elems per thread ----
    if (full) {
#pragma unroll
      for (int i = 0; i < (BM * BK / 8) / 256; ++i) {
        int idx = tid + i * 256;
        int row = idx >> 3;
        int kc  = (idx & 7) * 8;
        const float4* gp =
            (const float4*)(A + (size_t)(blockM + row) * K + (k0 + kc));
        float4 v0 = gp[0];
        float4 v1 = gp[1];
        uint4 pk;
        pk.x = pack2bf(v0.x, v0.y);
        pk.y = pack2bf(v0.z, v0.w);
        pk.z = pack2bf(v1.x, v1.y);
        pk.w = pack2bf(v1.z, v1.w);
        *(uint4*)&As[row * LDS_STRIDE + kc] = pk;
        if (i == 0)  // speculative prefetch of next K-step of the A stream
          __builtin_prefetch(A + (size_t)(blockM + row) * K + (k0 + BK + kc), 0, 1);
      }
    } else {
#pragma unroll
      for (int i = 0; i < (BM * BK / 8) / 256; ++i) {
        int idx = tid + i * 256;
        int row = idx >> 3;
        int kc  = (idx & 7) * 8;
        int gr  = blockM + row;
        uint4 pk = {0u, 0u, 0u, 0u};
        if (gr < M) {
          const float4* gp = (const float4*)(A + (size_t)gr * K + (k0 + kc));
          float4 v0 = gp[0];
          float4 v1 = gp[1];
          pk.x = pack2bf(v0.x, v0.y);
          pk.y = pack2bf(v0.z, v0.w);
          pk.z = pack2bf(v1.x, v1.y);
          pk.w = pack2bf(v1.z, v1.w);
        }
        *(uint4*)&As[row * LDS_STRIDE + kc] = pk;
      }
    }
    // ---- stage B (128x64 bf16, already N-major): 4 chunks per thread ----
#pragma unroll
    for (int i = 0; i < (BN * BK / 8) / 256; ++i) {
      int idx = tid + i * 256;
      int n  = idx >> 3;
      int kc = (idx & 7) * 8;
      const unsigned short* gp = Bt + (size_t)(blockN + n) * K + (k0 + kc);
      unsigned short* lp = &Bs[n * LDS_STRIDE + kc];
#if USE_ASYNC_B
      __builtin_amdgcn_global_load_async_to_lds_b128(
          (__attribute__((address_space(1))) v4i*)gp,
          (__attribute__((address_space(3))) v4i*)lp, 0, 0);
#else
      *(uint4*)lp = *(const uint4*)gp;
#endif
    }
#if USE_ASYNC_B
    __builtin_amdgcn_s_wait_asynccnt(0);
#endif
    __syncthreads();

    // ---- two 32-K chunks per staged K-step ----
#pragma unroll
    for (int c = 0; c < 2; ++c) {
      const int kb = c * 32;
      // A fragments (16x32): half=0 -> K kb+0..7 & kb+16..23, half=1 -> +8
      v16bf afr[2];
#pragma unroll
      for (int mt = 0; mt < 2; ++mt) {
        const unsigned short* base =
            &As[(32 * waveM + 16 * mt + l16) * LDS_STRIDE + kb];
        bf16x8 lo = *(const bf16x8*)(base + 8 * half);
        bf16x8 hi = *(const bf16x8*)(base + 16 + 8 * half);
#pragma unroll
        for (int q = 0; q < 8; ++q) { afr[mt][q] = lo[q]; afr[mt][8 + q] = hi[q]; }
      }
      // B fragments (32x16): lane n = l16, half=0 -> K kb+0..7 & kb+16..23, half=1 -> +8
      v16bf bfr[4];
#pragma unroll
      for (int nt = 0; nt < 4; ++nt) {
        const unsigned short* base =
            &Bs[(64 * waveN + 16 * nt + l16) * LDS_STRIDE + kb];
        bf16x8 lo = *(const bf16x8*)(base + 8 * half);
        bf16x8 hi = *(const bf16x8*)(base + 16 + 8 * half);
#pragma unroll
        for (int q = 0; q < 8; ++q) { bfr[nt][q] = lo[q]; bfr[nt][8 + q] = hi[q]; }
      }
#pragma unroll
      for (int mt = 0; mt < 2; ++mt)
#pragma unroll
        for (int nt = 0; nt < 4; ++nt)
          acc[mt][nt] = __builtin_amdgcn_wmma_f32_16x16x32_bf16(
              false, afr[mt], false, bfr[nt], (short)0, acc[mt][nt],
              false, false);
    }
    __syncthreads();
  }

  // ---- store D: VGPR r -> row (r + 8*half), col = lane&15 ----
#pragma unroll
  for (int mt = 0; mt < 2; ++mt) {
#pragma unroll
    for (int nt = 0; nt < 4; ++nt) {
      const int colBase = blockN + 64 * waveN + 16 * nt + l16;
#pragma unroll
      for (int r = 0; r < 8; ++r) {
        int row = blockM + 32 * waveM + 16 * mt + r + 8 * half;
        if (row < M) C[(size_t)row * N + colBase] = acc[mt][nt][r];
      }
    }
  }
}

// ---------------------------------------------------------------------------
// Elementwise / scatter kernels
// ---------------------------------------------------------------------------
__global__ void init_f32_kernel(float* __restrict__ p, float v, long long n) {
  long long i = (long long)blockIdx.x * blockDim.x + threadIdx.x;
  if (i < n) p[i] = v;
}

__global__ void deg_kernel(const long long* __restrict__ ei, float* __restrict__ deg,
                           long long E) {
  long long e = (long long)blockIdx.x * blockDim.x + threadIdx.x;
  if (e >= E) return;
  int c = (int)ei[E + e];
  atomicAdd(&deg[c], 1.0f);
}

__global__ void dinv_kernel(const float* __restrict__ deg, float* __restrict__ dinv,
                            long long n) {
  long long i = (long long)blockIdx.x * blockDim.x + threadIdx.x;
  if (i < n) {
    float d = deg[i];
    dinv[i] = (d > 0.0f) ? rsqrtf(d) : 0.0f;
  }
}

// scatter: hagg[col, f] += xW[row, f] * dinv[row]*dinv[col]; float4 per thread
__global__ void gcn_scatter_kernel(const long long* __restrict__ ei,
                                   const float* __restrict__ xW,
                                   const float* __restrict__ dinv,
                                   float* __restrict__ hagg, long long E) {
  long long idx = (long long)blockIdx.x * blockDim.x + threadIdx.x;
  long long e = idx >> 7;                 // H1/4 = 128 quads per edge
  int fq = (int)(idx & 127) * 4;
  if (e >= E) return;
  int r = (int)ei[e];
  int c = (int)ei[E + e];
  float norm = dinv[r] * dinv[c];
  float4 v = *(const float4*)(xW + (size_t)r * H1 + fq);
  float* dst = hagg + (size_t)c * H1 + fq;
  atomicAdd(dst + 0, v.x * norm);
  atomicAdd(dst + 1, v.y * norm);
  atomicAdd(dst + 2, v.z * norm);
  atomicAdd(dst + 3, v.w * norm);
}

// h = relu(hagg + self_loop + bias), in place
__global__ void finalize_gcn_kernel(float* __restrict__ hagg,
                                    const float* __restrict__ xW,
                                    const float* __restrict__ dinv,
                                    const float* __restrict__ b_gcn, long long nNodes) {
  long long idx = (long long)blockIdx.x * blockDim.x + threadIdx.x;
  long long n = idx >> 9;
  int f = (int)(idx & 511);
  if (n >= nNodes) return;
  float di = dinv[n];
  float v = hagg[idx] + xW[idx] * di * di + b_gcn[f];
  hagg[idx] = v > 0.0f ? v : 0.0f;
}

// per-node attention dots: a_src[n] = <g[n,:], att_src>, same for dst
__global__ void att_dot_kernel(const float* __restrict__ g,
                               const float* __restrict__ att_src,
                               const float* __restrict__ att_dst,
                               float* __restrict__ a_src, float* __restrict__ a_dst,
                               long long nNodes) {
  int wave = threadIdx.x >> 5;
  int lane = threadIdx.x & 31;
  long long node = (long long)blockIdx.x * 8 + wave;
  if (node >= nNodes) return;
  const float* gr = g + (size_t)node * C_HEAD;
  float s = 0.0f, d = 0.0f;
#pragma unroll
  for (int j = 0; j < C_HEAD / 32; ++j) {
    float v = gr[lane + 32 * j];
    s += v * att_src[lane + 32 * j];
    d += v * att_dst[lane + 32 * j];
  }
#pragma unroll
  for (int off = 16; off > 0; off >>= 1) {
    s += __shfl_xor(s, off, 32);
    d += __shfl_xor(d, off, 32);
  }
  if (lane == 0) { a_src[node] = s; a_dst[node] = d; }
}

// i in [0, E+N): e = leaky_relu(a_src[row]+a_dst[col], 0.2); segment max
__global__ void edge_logit_kernel(const long long* __restrict__ ei,
                                  const float* __restrict__ a_src,
                                  const float* __restrict__ a_dst,
                                  float* __restrict__ e_ws,
                                  unsigned* __restrict__ m_ord,
                                  long long E, long long nNodes) {
  long long i = (long long)blockIdx.x * blockDim.x + threadIdx.x;
  if (i >= E + nNodes) return;
  int r, c;
  if (i < E) { r = (int)ei[i]; c = (int)ei[E + i]; }
  else       { r = (int)(i - E); c = r; }
  float ev = a_src[r] + a_dst[c];
  ev = ev > 0.0f ? ev : 0.2f * ev;
  e_ws[i] = ev;
  atomicMax(&m_ord[c], f2ord(ev));
}

__global__ void edge_exp_kernel(const long long* __restrict__ ei,
                                const float* __restrict__ e_ws,
                                const unsigned* __restrict__ m_ord,
                                float* __restrict__ ex_ws,
                                float* __restrict__ denom,
                                long long E, long long nNodes) {
  long long i = (long long)blockIdx.x * blockDim.x + threadIdx.x;
  if (i >= E + nNodes) return;
  int c = (i < E) ? (int)ei[E + i] : (int)(i - E);
  float ex = __expf(e_ws[i] - ord2f(m_ord[c]));
  ex_ws[i] = ex;
  atomicAdd(&denom[c], ex);
}

// alpha = ex/denom[col]; out_agg[col,f] += g[row,f]*alpha; float4 per thread
__global__ void alpha_scatter_kernel(const long long* __restrict__ ei,
                                     const float* __restrict__ ex_ws,
                                     const float* __restrict__ denom,
                                     const float* __restrict__ g,
                                     float* __restrict__ out_agg,
                                     float* __restrict__ alpha_out,
                                     long long E, long long nNodes) {
  long long idx = (long long)blockIdx.x * blockDim.x + threadIdx.x;
  long long i = idx >> 6;                 // C_HEAD/4 = 64 quads per edge
  int fq = (int)(idx & 63) * 4;
  if (i >= E + nNodes) return;
  int r, c;
  if (i < E) { r = (int)ei[i]; c = (int)ei[E + i]; }
  else       { r = (int)(i - E); c = r; }
  float alpha = ex_ws[i] / denom[c];
  if (fq == 0) alpha_out[i] = alpha;
  float4 v = *(const float4*)(g + (size_t)r * C_HEAD + fq);
  float* dst = out_agg + (size_t)c * C_HEAD + fq;
  atomicAdd(dst + 0, v.x * alpha);
  atomicAdd(dst + 1, v.y * alpha);
  atomicAdd(dst + 2, v.z * alpha);
  atomicAdd(dst + 3, v.w * alpha);
}

__global__ void finalize_out_kernel(const float* __restrict__ out_agg,
                                    const float* __restrict__ b_gat,
                                    float* __restrict__ out, long long nNodes) {
  long long idx = (long long)blockIdx.x * blockDim.x + threadIdx.x;
  long long n = idx >> 8;
  int f = (int)(idx & 255);
  if (n >= nNodes) return;
  out[idx] = out_agg[idx] + b_gat[f];
}

__global__ void write_ei_kernel(const long long* __restrict__ ei,
                                float* __restrict__ ei_out,
                                long long E, long long nNodes) {
  long long i = (long long)blockIdx.x * blockDim.x + threadIdx.x;
  long long T = E + nNodes;
  if (i >= T) return;
  long long r, c;
  if (i < E) { r = ei[i]; c = ei[E + i]; }
  else       { r = i - E; c = r; }
  ei_out[i]     = (float)r;
  ei_out[T + i] = (float)c;
}

// ---------------------------------------------------------------------------
// Launcher
// ---------------------------------------------------------------------------
extern "C" void kernel_launch(void* const* d_in, const int* in_sizes, int n_in,
                              void* d_out, int out_size, void* d_ws, size_t ws_size,
                              hipStream_t stream) {
  const float*     x       = (const float*)d_in[0];
  const long long* ei      = (const long long*)d_in[1];   // int64 [2, E]
  const float*     W_gcn   = (const float*)d_in[2];
  const float*     b_gcn   = (const float*)d_in[3];
  const float*     W_gat   = (const float*)d_in[4];
  const float*     att_src = (const float*)d_in[5];
  const float*     att_dst = (const float*)d_in[6];
  const float*     b_gat   = (const float*)d_in[7];
  float*           out     = (float*)d_out;

  const long long N = in_sizes[0] / F_IN;   // 20000
  const long long E = in_sizes[1] / 2;      // 320000
  const long long T = E + N;                // edges + self loops

  // ---- workspace carving (all chunks keep 16B alignment) ----
  float* ws = (float*)d_ws;
  float* xW    = ws;                 ws += (size_t)N * H1;   // x@W_gcn; reused as out_agg
  float* hagg  = ws;                 ws += (size_t)N * H1;   // gcn agg -> h (in place)
  float* gbuf  = ws;                 ws += (size_t)N * C_HEAD;
  float* deg   = ws;                 ws += N;
  float* dinv  = ws;                 ws += N;
  float* a_src = ws;                 ws += N;
  float* a_dst = ws;                 ws += N;
  float* denom = ws;                 ws += N;
  unsigned* m_ord = (unsigned*)ws;   ws += N;
  float* e_ws  = ws;                 ws += T;
  float* ex_ws = ws;                 ws += T;
  unsigned short* Wgcn_t = (unsigned short*)ws;             // [H1][F_IN] bf16, 2MB
  ws += (size_t)F_IN * H1 / 2;
  unsigned short* Wgat_t = (unsigned short*)ws;             // [C_HEAD][H1] bf16
  ws += (size_t)H1 * C_HEAD / 2;
  float* out_agg = xW;               // xW dead after finalize_gcn

  const int B = 256;
  auto blocks = [](long long n, int b) { return (unsigned)((n + b - 1) / b); };

  // ---- weight pre-pass: fp32 [K][N] -> bf16 [N][K] ----
  convert_transpose_kernel<<<blocks((long long)F_IN * H1, B), B, 0, stream>>>(
      W_gcn, Wgcn_t, F_IN, H1);
  convert_transpose_kernel<<<blocks((long long)H1 * C_HEAD, B), B, 0, stream>>>(
      W_gat, Wgat_t, H1, C_HEAD);

  // ---- GCN layer ----
  init_f32_kernel<<<blocks(N, B), B, 0, stream>>>(deg, 1.0f, N);     // self loop
  deg_kernel<<<blocks(E, B), B, 0, stream>>>(ei, deg, E);
  dinv_kernel<<<blocks(N, B), B, 0, stream>>>(deg, dinv, N);

  {
    dim3 grid(H1 / BN, blocks(N, BM));
    gemm_bf16_wmma_kernel<<<grid, B, 0, stream>>>(x, Wgcn_t, xW, (int)N, H1, F_IN);
  }
  init_f32_kernel<<<blocks(N * H1, B), B, 0, stream>>>(hagg, 0.0f, N * H1);
  gcn_scatter_kernel<<<blocks(E * (H1 / 4), B), B, 0, stream>>>(ei, xW, dinv, hagg, E);
  finalize_gcn_kernel<<<blocks(N * H1, B), B, 0, stream>>>(hagg, xW, dinv, b_gcn, N);

  // ---- GAT layer ----
  {
    dim3 grid(C_HEAD / BN, blocks(N, BM));
    gemm_bf16_wmma_kernel<<<grid, B, 0, stream>>>(hagg, Wgat_t, gbuf, (int)N, C_HEAD, H1);
  }
  att_dot_kernel<<<blocks(N, 8), B, 0, stream>>>(gbuf, att_src, att_dst,
                                                 a_src, a_dst, N);

  init_f32_kernel<<<blocks(N, B), B, 0, stream>>>((float*)m_ord, 0.0f, N); // ord min
  init_f32_kernel<<<blocks(N, B), B, 0, stream>>>(denom, 0.0f, N);
  edge_logit_kernel<<<blocks(T, B), B, 0, stream>>>(ei, a_src, a_dst, e_ws, m_ord, E, N);
  edge_exp_kernel<<<blocks(T, B), B, 0, stream>>>(ei, e_ws, m_ord, ex_ws, denom, E, N);

  init_f32_kernel<<<blocks(N * C_HEAD, B), B, 0, stream>>>(out_agg, 0.0f, N * C_HEAD);
  float* alpha_out = out + (size_t)N * C_HEAD + 2 * (size_t)T;
  alpha_scatter_kernel<<<blocks(T * (C_HEAD / 4), B), B, 0, stream>>>(
      ei, ex_ws, denom, gbuf, out_agg, alpha_out, E, N);

  finalize_out_kernel<<<blocks(N * C_HEAD, B), B, 0, stream>>>(out_agg, b_gat, out, N);
  write_ei_kernel<<<blocks(T, B), B, 0, stream>>>(ei, out + (size_t)N * C_HEAD, E, N);
}
